// BigAttn_48309792145810
// MI455X (gfx1250) — matrix operand order repaired and checked by reading, not compiled, not run on hardware
//
#include <hip/hip_runtime.h>
#include <math.h>

// ---------------- problem constants (match reference) ----------------
#define NNODES 20000
#define NEDGE  160000
#define NSUB   25
#define NHD    2
#define NC     16
#define HC     32            // NHD*NC
#define FIN    5
#define FCW    800
#define ETMAX  (NEDGE + NNODES)   // layer-2 edge count incl. self loops

typedef __attribute__((ext_vector_type(16))) _Float16 v16h;
typedef __attribute__((ext_vector_type(8)))  _Float16 v8h;
typedef __attribute__((ext_vector_type(8)))  float    v8f;

static __device__ __forceinline__ void atomicMaxF(float* addr, float v) {
  int* ia = (int*)addr;
  int old = __float_as_int(*addr);
  while (__int_as_float(old) < v) {
    int assumed = old;
    old = atomicCAS(ia, assumed, __float_as_int(v));
    if (old == assumed) break;
  }
}

// ---------------- utility ----------------
__global__ void k_fill(float* p, float v, long n) {
  long i = blockIdx.x * (long)blockDim.x + threadIdx.x;
  if (i < n) p[i] = v;
}

__global__ void k_ew_mean(const float* __restrict__ ew, float* __restrict__ ewmean) {
  __shared__ float sm[256];
  int s = blockIdx.x;
  float acc = 0.f;
  for (int e = threadIdx.x; e < NEDGE; e += 256) acc += ew[(long)s * NEDGE + e];
  sm[threadIdx.x] = acc; __syncthreads();
  for (int off = 128; off > 0; off >>= 1) {
    if ((int)threadIdx.x < off) sm[threadIdx.x] += sm[threadIdx.x + off];
    __syncthreads();
  }
  if (threadIdx.x == 0) ewmean[s] = sm[0] / (float)NEDGE;
}

// ---------------- GATv2 node transform: xl = x@Wl+bl, xr = x@Wr+br ----------------
__global__ void __launch_bounds__(256)
k_node_transform(const float* __restrict__ xin, int fin, int shared_x,
                 const float* __restrict__ Wl, const float* __restrict__ bl,
                 const float* __restrict__ Wr, const float* __restrict__ br,
                 float* __restrict__ XL, float* __restrict__ XR) {
  long t = blockIdx.x * (long)blockDim.x + threadIdx.x;
  if (t >= (long)NSUB * NNODES) return;
  int s = (int)(t / NNODES), n = (int)(t % NNODES);
  const float* xp = xin + (shared_x ? (long)n * fin : ((long)s * NNODES + n) * fin);
  float ol[HC], orr[HC];
  #pragma unroll
  for (int j = 0; j < HC; j++) { ol[j] = bl[s * HC + j]; orr[j] = br[s * HC + j]; }
  for (int k = 0; k < fin; k++) {
    float xv = xp[k];
    const float* wlk = Wl + ((long)s * fin + k) * HC;
    const float* wrk = Wr + ((long)s * fin + k) * HC;
    #pragma unroll
    for (int j = 0; j < HC; j++) { ol[j] += xv * wlk[j]; orr[j] += xv * wrk[j]; }
  }
  float* xlo = XL + ((long)s * NNODES + n) * HC;
  float* xro = XR + ((long)s * NNODES + n) * HC;
  #pragma unroll
  for (int j = 0; j < HC; j++) { xlo[j] = ol[j]; xro[j] = orr[j]; }
}

// ---------------- edge pass A: scores + segment max + degree ----------------
__global__ void __launch_bounds__(256)
k_edge_score(const int* __restrict__ ei, const float* __restrict__ ew,
             const float* __restrict__ ewmean, const float* __restrict__ We,
             const float* __restrict__ att, const float* __restrict__ XL,
             const float* __restrict__ XR, float* __restrict__ ESC,
             float* __restrict__ SMAX, float* __restrict__ DEG, int ET) {
  long t = blockIdx.x * (long)blockDim.x + threadIdx.x;
  if (t >= (long)NSUB * ET) return;
  int s = (int)(t / ET), e = (int)(t % ET);
  int src, dst; float w;
  if (e < NEDGE) {
    src = ei[((long)s * 2 + 0) * NEDGE + e];
    dst = ei[((long)s * 2 + 1) * NEDGE + e];
    w   = ew[(long)s * NEDGE + e];
  } else {                       // self loop with fill_value = mean(edge_attr)
    src = dst = e - NEDGE;
    w   = ewmean[s];
  }
  const float* xl = XL + ((long)s * NNODES + src) * HC;
  const float* xr = XR + ((long)s * NNODES + dst) * HC;
  float sc0 = 0.f, sc1 = 0.f;
  #pragma unroll
  for (int j = 0; j < HC; j++) {
    float m = xl[j] + xr[j] + w * We[s * HC + j];
    m = m > 0.f ? m : 0.2f * m;                        // leaky(0.2)
    float a = att[((long)s * NHD + (j >> 4)) * NC + (j & 15)];
    if (j < NC) sc0 += m * a; else sc1 += m * a;
  }
  float* esc = ESC + ((long)s * ETMAX + e) * 2;
  esc[0] = sc0; esc[1] = sc1;
  atomicMaxF(SMAX + ((long)s * NNODES + dst) * 2 + 0, sc0);
  atomicMaxF(SMAX + ((long)s * NNODES + dst) * 2 + 1, sc1);
  atomicAdd(DEG + (long)s * NNODES + dst, 1.0f);
}

// ---------------- edge pass B: exp + denominator ----------------
__global__ void __launch_bounds__(256)
k_edge_exp(const int* __restrict__ ei, float* __restrict__ ESC,
           const float* __restrict__ SMAX, float* __restrict__ DEN, int ET) {
  long t = blockIdx.x * (long)blockDim.x + threadIdx.x;
  if (t >= (long)NSUB * ET) return;
  int s = (int)(t / ET), e = (int)(t % ET);
  int dst = (e < NEDGE) ? ei[((long)s * 2 + 1) * NEDGE + e] : (e - NEDGE);
  float* esc = ESC + ((long)s * ETMAX + e) * 2;
  const float* smax = SMAX + ((long)s * NNODES + dst) * 2;
  float e0 = __expf(esc[0] - smax[0]);
  float e1 = __expf(esc[1] - smax[1]);
  esc[0] = e0; esc[1] = e1;
  atomicAdd(DEN + ((long)s * NNODES + dst) * 2 + 0, e0);
  atomicAdd(DEN + ((long)s * NNODES + dst) * 2 + 1, e1);
}

// ---------------- edge pass C: alpha-weighted scatter add ----------------
__global__ void __launch_bounds__(256)
k_edge_aggr(const int* __restrict__ ei, const float* __restrict__ ESC,
            const float* __restrict__ DEN, const float* __restrict__ XL,
            float* __restrict__ ACC, int ET) {
  long t = blockIdx.x * (long)blockDim.x + threadIdx.x;
  if (t >= (long)NSUB * ET) return;
  int s = (int)(t / ET), e = (int)(t % ET);
  int src, dst;
  if (e < NEDGE) {
    src = ei[((long)s * 2 + 0) * NEDGE + e];
    dst = ei[((long)s * 2 + 1) * NEDGE + e];
  } else { src = dst = e - NEDGE; }
  const float* esc = ESC + ((long)s * ETMAX + e) * 2;
  const float* den = DEN + ((long)s * NNODES + dst) * 2;
  float a0 = esc[0] / (den[0] + 1e-16f);
  float a1 = esc[1] / (den[1] + 1e-16f);
  const float* xl = XL + ((long)s * NNODES + src) * HC;
  float* acc = ACC + ((long)s * NNODES + dst) * HC;
  #pragma unroll
  for (int j = 0; j < HC; j++) atomicAdd(acc + j, xl[j] * (j < NC ? a0 : a1));
}

// ---------------- finalize: deg-mean (optional), head mean, + bias ----------------
__global__ void __launch_bounds__(256)
k_finalize(const float* __restrict__ ACC, const float* __restrict__ DEG,
           const float* __restrict__ bias, float* __restrict__ H, int aggr_mean) {
  long t = blockIdx.x * (long)blockDim.x + threadIdx.x;
  if (t >= (long)NSUB * NNODES) return;
  int s = (int)(t / NNODES), n = (int)(t % NNODES);
  const float* acc = ACC + ((long)s * NNODES + n) * HC;
  float inv = aggr_mean ? (1.f / fmaxf(DEG[(long)s * NNODES + n], 1.f)) : 1.f;
  float* h = H + ((long)s * NNODES + n) * NC;
  #pragma unroll
  for (int c = 0; c < NC; c++)
    h[c] = 0.5f * (acc[c] + acc[NC + c]) * inv + bias[s * NC + c];
}

// ---------------- BatchNorm over nodes per (s,c) ----------------
__global__ void k_bn_stats(const float* __restrict__ H, float* __restrict__ STAT) {
  int s = blockIdx.x / NC, c = blockIdx.x % NC;
  __shared__ float s1[256], s2[256];
  float a = 0.f, b = 0.f;
  for (int n = threadIdx.x; n < NNODES; n += 256) {
    float v = H[((long)s * NNODES + n) * NC + c];
    a += v; b += v * v;
  }
  s1[threadIdx.x] = a; s2[threadIdx.x] = b; __syncthreads();
  for (int off = 128; off > 0; off >>= 1) {
    if ((int)threadIdx.x < off) { s1[threadIdx.x] += s1[threadIdx.x + off]; s2[threadIdx.x] += s2[threadIdx.x + off]; }
    __syncthreads();
  }
  if (threadIdx.x == 0) {
    float m = s1[0] / (float)NNODES;
    STAT[blockIdx.x * 2 + 0] = m;
    STAT[blockIdx.x * 2 + 1] = s2[0] / (float)NNODES - m * m;   // biased var
  }
}

__global__ void __launch_bounds__(256)
k_bn_apply(float* __restrict__ H, const float* __restrict__ STAT,
           const float* __restrict__ g, const float* __restrict__ b, float slope) {
  long t = blockIdx.x * (long)blockDim.x + threadIdx.x;
  if (t >= (long)NSUB * NNODES * NC) return;
  int c = (int)(t % NC);
  int s = (int)(t / ((long)NNODES * NC));
  int sc = s * NC + c;
  float x = (H[t] - STAT[sc * 2]) * rsqrtf(STAT[sc * 2 + 1] + 1e-5f) * g[sc] + b[sc];
  H[t] = x > 0.f ? x : slope * x;
}

// ---------------- concat: feature = s*32 + branch*16 + c ----------------
__global__ void __launch_bounds__(256)
k_concat(const float* __restrict__ HA, const float* __restrict__ HB, float* __restrict__ HCAT) {
  long t = blockIdx.x * (long)blockDim.x + threadIdx.x;
  if (t >= (long)NSUB * NNODES * NC) return;
  int c = (int)(t % NC);
  int n = (int)((t / NC) % NNODES);
  int s = (int)(t / ((long)NNODES * NC));
  HCAT[(long)n * FCW + s * 2 * NC + c]      = HA[t];
  HCAT[(long)n * FCW + s * 2 * NC + NC + c] = HB[t];
}

// ---------------- f32 -> f16 conversions ----------------
__global__ void k_cvt_f16(const float* __restrict__ src, _Float16* __restrict__ dst, long n) {
  long i = blockIdx.x * (long)blockDim.x + threadIdx.x;
  if (i < n) dst[i] = (_Float16)src[i];
}
__global__ void k_cvt_wT(const float* __restrict__ w, _Float16* __restrict__ wT, int K, int Ncol) {
  long t = blockIdx.x * (long)blockDim.x + threadIdx.x;
  if (t >= (long)K * Ncol) return;
  int k = (int)(t / Ncol), n = (int)(t % Ncol);
  wT[(long)n * K + k] = (_Float16)w[t];                 // column-major for contiguous B loads
}

// ---------------- WMMA GEMM, 2Mx2N register-blocked --------------------
// C[M,N] = A[M,K](f16 row-major) * B[K,N](f16, stored column-major) + bias
// One wave owns a 32x32 output tile = 4 WMMA accumulators; per K-step it
// loads 2 A fragments + 2 B fragments (8x b128) and issues 4 WMMAs ->
// 1 KB of operand traffic per WMMA (16 FLOP/byte) instead of 2 KB.
__global__ void __launch_bounds__(256)
k_gemm_wmma(const _Float16* __restrict__ A, const _Float16* __restrict__ BT,
            const float* __restrict__ bias, float* __restrict__ C,
            int M, int K, int Ncol) {
  int gwave = (int)((blockIdx.x * 256u + threadIdx.x) >> 5);
  int lane  = threadIdx.x & 31;
  int nGrp = Ncol >> 5, mGrp = M >> 5;          // 32x32 groups
  int mG = gwave / nGrp, nG = gwave % nGrp;
  if (mG >= mGrp) return;                        // wave-uniform: EXEC stays all-ones
  int hi = lane >> 4;                            // lane half selects K sub-ranges (ISA 7.12.2)
  long rowA0 = (long)mG * 32 + (lane & 15);
  long colB0 = (long)nG * 32 + (lane & 15);
  const _Float16* aB0 = A  + rowA0 * K + hi * 8;        // A: K[hi*8..+7], K[16+hi*8..+7]
  const _Float16* aB1 = aB0 + 16 * (long)K;
  const _Float16* bB0 = BT + colB0 * K + hi * 8;        // B: K[hi*8..+7], K[16+hi*8..+7]
  const _Float16* bB1 = bB0 + 16 * (long)K;
  v8f acc00 = {}, acc01 = {}, acc10 = {}, acc11 = {};
  for (int k0 = 0; k0 < K; k0 += 32) {
    __builtin_prefetch(aB0 + k0 + 128, 0, 0);
    __builtin_prefetch(bB0 + k0 + 128, 0, 0);
    v8h a00 = *(const v8h*)(aB0 + k0);
    v8h a01 = *(const v8h*)(aB0 + k0 + 16);
    v8h a10 = *(const v8h*)(aB1 + k0);
    v8h a11 = *(const v8h*)(aB1 + k0 + 16);
    v8h b00 = *(const v8h*)(bB0 + k0);
    v8h b01 = *(const v8h*)(bB0 + k0 + 16);
    v8h b10 = *(const v8h*)(bB1 + k0);
    v8h b11 = *(const v8h*)(bB1 + k0 + 16);
    v16h a0, a1, b0, b1;
    #pragma unroll
    for (int i = 0; i < 8; i++) {
      a0[i] = a00[i]; a0[8 + i] = a01[i];
      a1[i] = a10[i]; a1[8 + i] = a11[i];
      b0[i] = b00[i]; b0[8 + i] = b01[i];
      b1[i] = b10[i]; b1[8 + i] = b11[i];
    }
    acc00 = __builtin_amdgcn_wmma_f32_16x16x32_f16(false, a0, false, b0, (short)0, acc00, false, false);
    acc01 = __builtin_amdgcn_wmma_f32_16x16x32_f16(false, a0, false, b1, (short)0, acc01, false, false);
    acc10 = __builtin_amdgcn_wmma_f32_16x16x32_f16(false, a1, false, b0, (short)0, acc10, false, false);
    acc11 = __builtin_amdgcn_wmma_f32_16x16x32_f16(false, a1, false, b1, (short)0, acc11, false, false);
  }
  // C/D layout: lanes 0-15 -> rows 0-7 in vgprs 0-7, lanes 16-31 -> rows 8-15
  int col0 = nG * 32 + (lane & 15), col1 = col0 + 16;
  long rb0 = (long)mG * 32 + hi * 8, rb1 = rb0 + 16;
  float bv0 = bias[col0], bv1 = bias[col1];
  #pragma unroll
  for (int i = 0; i < 8; i++) {
    C[(rb0 + i) * (long)Ncol + col0] = acc00[i] + bv0;
    C[(rb0 + i) * (long)Ncol + col1] = acc01[i] + bv1;
    C[(rb1 + i) * (long)Ncol + col0] = acc10[i] + bv0;
    C[(rb1 + i) * (long)Ncol + col1] = acc11[i] + bv1;
  }
}

// ---------------- FC BatchNorm ----------------
__global__ void k_fc_stats(const float* __restrict__ Z, float* __restrict__ STAT) {
  int c = blockIdx.x;                      // 0..FCW-1
  __shared__ float s1[256], s2[256];
  float a = 0.f, b = 0.f;
  for (int n = threadIdx.x; n < NNODES; n += 256) {
    float v = Z[(long)n * FCW + c];
    a += v; b += v * v;
  }
  s1[threadIdx.x] = a; s2[threadIdx.x] = b; __syncthreads();
  for (int off = 128; off > 0; off >>= 1) {
    if ((int)threadIdx.x < off) { s1[threadIdx.x] += s1[threadIdx.x + off]; s2[threadIdx.x] += s2[threadIdx.x + off]; }
    __syncthreads();
  }
  if (threadIdx.x == 0) {
    float m = s1[0] / (float)NNODES;
    STAT[c * 2 + 0] = m;
    STAT[c * 2 + 1] = s2[0] / (float)NNODES - m * m;
  }
}

__global__ void __launch_bounds__(256)
k_fc_bn_relu(float* __restrict__ Z, const float* __restrict__ STAT,
             const float* __restrict__ g, const float* __restrict__ bt) {
  long t = blockIdx.x * (long)blockDim.x + threadIdx.x;
  if (t >= (long)NNODES * FCW) return;
  int c = (int)(t % FCW);
  float x = (Z[t] - STAT[c * 2]) * rsqrtf(STAT[c * 2 + 1] + 1e-5f) * g[c] + bt[c];
  Z[t] = fmaxf(x, 0.f);
}

// ---------------- final tiny projection [800 -> 5] ----------------
__global__ void __launch_bounds__(256)
k_fc3(const float* __restrict__ Z, const float* __restrict__ w3,
      const float* __restrict__ b3, float* __restrict__ out) {
  int n = blockIdx.x * blockDim.x + threadIdx.x;
  if (n >= NNODES) return;
  float acc[5];
  #pragma unroll
  for (int j = 0; j < 5; j++) acc[j] = b3[j];
  for (int k = 0; k < FCW; k++) {
    float z = Z[(long)n * FCW + k];
    #pragma unroll
    for (int j = 0; j < 5; j++) acc[j] += z * w3[k * 5 + j];
  }
  #pragma unroll
  for (int j = 0; j < 5; j++) out[(long)n * 5 + j] = acc[j];
}

// =====================================================================
static inline dim3 gridFor(long n) { return dim3((unsigned)((n + 255) / 256)); }

struct GatP { const float *Wl, *bl, *Wr, *br, *We, *att, *bias; };

static void run_layer(hipStream_t st, const float* xin, int fin, int shared_x,
                      const GatP& p, const int* ei, const float* ew, const float* ewm,
                      float* XL, float* XR, float* SMAX, float* DEN, float* DEG,
                      float* ESC, float* Hout, int ET, int aggr_mean) {
  long SN = (long)NSUB * NNODES;
  long SE = (long)NSUB * ET;
  k_node_transform<<<gridFor(SN), 256, 0, st>>>(xin, fin, shared_x, p.Wl, p.bl, p.Wr, p.br, XL, XR);
  k_fill<<<gridFor(SN * 2), 256, 0, st>>>(SMAX, -3.0e38f, SN * 2);
  k_fill<<<gridFor(SN * 2), 256, 0, st>>>(DEN, 0.f, SN * 2);
  k_fill<<<gridFor(SN), 256, 0, st>>>(DEG, 0.f, SN);
  k_edge_score<<<gridFor(SE), 256, 0, st>>>(ei, ew, ewm, p.We, p.att, XL, XR, ESC, SMAX, DEG, ET);
  k_fill<<<gridFor(SN * HC), 256, 0, st>>>(XR, 0.f, SN * HC);   // reuse XR as scatter accumulator
  k_edge_exp<<<gridFor(SE), 256, 0, st>>>(ei, ESC, SMAX, DEN, ET);
  k_edge_aggr<<<gridFor(SE), 256, 0, st>>>(ei, ESC, DEN, XL, XR, ET);
  k_finalize<<<gridFor(SN), 256, 0, st>>>(XR, DEG, p.bias, Hout, aggr_mean);
}

extern "C" void kernel_launch(void* const* d_in, const int* in_sizes, int n_in,
                              void* d_out, int out_size, void* d_ws, size_t ws_size,
                              hipStream_t stream) {
  (void)in_sizes; (void)n_in; (void)out_size; (void)ws_size;
  // -------- input flattening (setup_inputs dict order, nested dicts in insertion order) ----
  const float* x_input = (const float*)d_in[0];
  // d_in[1] bf, d_in[4..5] train_* : unused by forward
  const int*   sub_ei  = (const int*)d_in[2];       // [S,2,E]
  const float* sub_ew  = (const float*)d_in[3];     // [S,E]
  auto gp = [&](int base) {
    GatP p; p.Wl = (const float*)d_in[base + 0]; p.bl  = (const float*)d_in[base + 1];
            p.Wr = (const float*)d_in[base + 2]; p.br  = (const float*)d_in[base + 3];
            p.We = (const float*)d_in[base + 4]; p.att = (const float*)d_in[base + 5];
            p.bias = (const float*)d_in[base + 6]; return p;
  };
  GatP g1a = gp(6), g1b = gp(13), g2a = gp(20), g2b = gp(27);
  const float* bn1a_g = (const float*)d_in[34]; const float* bn1a_b = (const float*)d_in[35];
  const float* bn1b_g = (const float*)d_in[36]; const float* bn1b_b = (const float*)d_in[37];
  const float* bn2a_g = (const float*)d_in[38]; const float* bn2a_b = (const float*)d_in[39];
  const float* bn2b_g = (const float*)d_in[40]; const float* bn2b_b = (const float*)d_in[41];
  const float* w1 = (const float*)d_in[42]; const float* b1 = (const float*)d_in[43];
  const float* fg1 = (const float*)d_in[44]; const float* fb1 = (const float*)d_in[45];
  const float* w2 = (const float*)d_in[46]; const float* b2 = (const float*)d_in[47];
  const float* fg2 = (const float*)d_in[48]; const float* fb2 = (const float*)d_in[49];
  const float* w3 = (const float*)d_in[50]; const float* b3 = (const float*)d_in[51];
  float* out = (float*)d_out;

  // -------- workspace carve-up (floats) --------
  float* ws = (float*)d_ws;  size_t o = 0;
  float* XL   = ws + o; o += (size_t)NSUB * NNODES * HC;
  float* XR   = ws + o; o += (size_t)NSUB * NNODES * HC;     // doubles as scatter accumulator
  float* SMAX = ws + o; o += (size_t)NSUB * NNODES * 2;
  float* DEN  = ws + o; o += (size_t)NSUB * NNODES * 2;
  float* DEG  = ws + o; o += (size_t)NSUB * NNODES;
  float* ESC  = ws + o; o += (size_t)NSUB * ETMAX * 2;
  float* H1   = ws + o; o += (size_t)NSUB * NNODES * NC;
  float* HA   = ws + o; o += (size_t)NSUB * NNODES * NC;
  float* HB   = ws + o; o += (size_t)NSUB * NNODES * NC;
  float* HCAT = ws + o; o += (size_t)NNODES * FCW;
  float* Z1   = ws + o; o += (size_t)NNODES * FCW;
  float* Z2   = ws + o; o += (size_t)NNODES * FCW;
  float* EWM  = ws + o; o += 32;
  float* BNST = ws + o; o += (size_t)NSUB * NC * 2;
  float* FST  = ws + o; o += (size_t)FCW * 2;
  _Float16* AH = (_Float16*)(ws + o); o += (size_t)NNODES * FCW / 2;
  _Float16* WH = (_Float16*)(ws + o); o += (size_t)FCW * FCW / 2;

  long SNC = (long)NSUB * NNODES * NC;

  k_ew_mean<<<NSUB, 256, 0, stream>>>(sub_ew, EWM);

  // -------- branch A (aggr='mean') --------
  run_layer(stream, x_input, FIN, 1, g1a, sub_ei, sub_ew, EWM, XL, XR, SMAX, DEN, DEG, ESC, H1, NEDGE, 1);
  k_bn_stats<<<NSUB * NC, 256, 0, stream>>>(H1, BNST);
  k_bn_apply<<<gridFor(SNC), 256, 0, stream>>>(H1, BNST, bn1a_g, bn1a_b, 0.01f);
  run_layer(stream, H1, NC, 0, g2a, sub_ei, sub_ew, EWM, XL, XR, SMAX, DEN, DEG, ESC, HA, ETMAX, 1);
  k_bn_stats<<<NSUB * NC, 256, 0, stream>>>(HA, BNST);
  k_bn_apply<<<gridFor(SNC), 256, 0, stream>>>(HA, BNST, bn2a_g, bn2a_b, 0.01f);

  // -------- branch B (aggr='add') --------
  run_layer(stream, x_input, FIN, 1, g1b, sub_ei, sub_ew, EWM, XL, XR, SMAX, DEN, DEG, ESC, H1, NEDGE, 0);
  k_bn_stats<<<NSUB * NC, 256, 0, stream>>>(H1, BNST);
  k_bn_apply<<<gridFor(SNC), 256, 0, stream>>>(H1, BNST, bn1b_g, bn1b_b, 0.01f);
  run_layer(stream, H1, NC, 0, g2b, sub_ei, sub_ew, EWM, XL, XR, SMAX, DEN, DEG, ESC, HB, ETMAX, 0);
  k_bn_stats<<<NSUB * NC, 256, 0, stream>>>(HB, BNST);
  k_bn_apply<<<gridFor(SNC), 256, 0, stream>>>(HB, BNST, bn2b_g, bn2b_b, 0.01f);

  // -------- concat + FC head (WMMA f16, 32x32 tiles per wave) --------
  k_concat<<<gridFor(SNC), 256, 0, stream>>>(HA, HB, HCAT);

  long gemmWaves = (long)(NNODES / 32) * (FCW / 32);     // 625 * 25
  dim3 gemmGrid((unsigned)((gemmWaves * 32 + 255) / 256));

  // FC1
  k_cvt_f16<<<gridFor((long)NNODES * FCW), 256, 0, stream>>>(HCAT, AH, (long)NNODES * FCW);
  k_cvt_wT<<<gridFor((long)FCW * FCW), 256, 0, stream>>>(w1, WH, FCW, FCW);
  k_gemm_wmma<<<gemmGrid, 256, 0, stream>>>(AH, WH, b1, Z1, NNODES, FCW, FCW);
  k_fc_stats<<<FCW, 256, 0, stream>>>(Z1, FST);
  k_fc_bn_relu<<<gridFor((long)NNODES * FCW), 256, 0, stream>>>(Z1, FST, fg1, fb1);

  // FC2
  k_cvt_f16<<<gridFor((long)NNODES * FCW), 256, 0, stream>>>(Z1, AH, (long)NNODES * FCW);
  k_cvt_wT<<<gridFor((long)FCW * FCW), 256, 0, stream>>>(w2, WH, FCW, FCW);
  k_gemm_wmma<<<gemmGrid, 256, 0, stream>>>(AH, WH, b2, Z2, NNODES, FCW, FCW);
  k_fc_stats<<<FCW, 256, 0, stream>>>(Z2, FST);
  k_fc_bn_relu<<<gridFor((long)NNODES * FCW), 256, 0, stream>>>(Z2, FST, fg2, fb2);

  // FC3
  k_fc3<<<gridFor(NNODES), 256, 0, stream>>>(Z2, w3, b3, out);
}
